// MultiHeadAttention_9826885173907
// MI455X (gfx1250) — matrix operand order repaired and checked by reading, not compiled, not run on hardware
//
#include <hip/hip_runtime.h>
#include <math.h>

// Problem constants (B=2, T=2048, C=2048, H=16, D=128)
#define BB 2
#define TT 2048
#define CC 2048
#define HH 16
#define DD 128
#define MROWS (BB * TT)   // 4096
#define NQKV  (3 * CC)    // 6144

typedef __bf16 bf16_t;
typedef __attribute__((ext_vector_type(16))) __bf16 v16bf;
typedef __attribute__((ext_vector_type(4)))  __bf16 v4bf;
typedef __attribute__((ext_vector_type(8)))  float  v8f;

// ---------------------------------------------------------------------------
// CDNA5 async global->LDS copy (bypasses VGPRs, tracked by ASYNCcnt).
// dst: generic pointer into __shared__ (low 32 bits = LDS byte address).
// ---------------------------------------------------------------------------
__device__ __forceinline__ void async_ld_b128(void* lds_dst, const void* gsrc) {
  uint32_t l = (uint32_t)(uintptr_t)lds_dst;
  asm volatile("global_load_async_to_lds_b128 %0, %1, off"
               :: "v"(l), "v"(gsrc)
               : "memory");
}

#if __has_builtin(__builtin_amdgcn_s_wait_asynccnt)
#define WAIT_ASYNC0() __builtin_amdgcn_s_wait_asynccnt(0)
#else
#define WAIT_ASYNC0() asm volatile("s_wait_asynccnt 0x0" ::: "memory")
#endif

// ---------------------------------------------------------------------------
// WMMA helper: D(16x16,f32) = A(16x32,bf16) * B(32x16,bf16) + C
// ---------------------------------------------------------------------------
__device__ __forceinline__ v8f wmma_bf16f32(v16bf a, v16bf b, v8f c) {
  return __builtin_amdgcn_wmma_f32_16x16x32_bf16(
      /*neg_a=*/false, a, /*neg_b=*/false, b,
      /*c_mod=*/(short)0, c, /*reuse_a=*/false, /*reuse_b=*/false);
}

// A fragment (16x32 bf16), source row-major with `stride` elements per row.
// Layout used here (contiguous k, same as B): lane&15 = row M; lanes 0-15 hold K 0..15, lanes 16-31 K 16..31.
__device__ __forceinline__ v16bf load_a_frag(const bf16_t* base, int stride) {
  const int lane = threadIdx.x & 31;
  const int row  = lane & 15;
  const int kb   = (lane >> 4) << 4;  // 0 or 16
  const bf16_t* r = base + row * stride;
  v16bf a;
#pragma unroll
  for (int i = 0; i < 16; ++i) a[i] = r[kb + i];
  return a;
}

// B fragment (32x16 bf16) from an N-major buffer: baseT[n * stride + k].
// Layout used here (contiguous k, same as A): lane&15 = column N; lanes 0-15 hold K 0..15, lanes 16-31 K 16..31.
__device__ __forceinline__ v16bf load_b_fragT(const bf16_t* baseT, int stride) {
  const int lane = threadIdx.x & 31;
  const int col  = lane & 15;
  const int ks   = (lane >> 4) << 4;  // 0 or 16
  const bf16_t* r = baseT + col * stride + ks;
  v16bf b;
#pragma unroll
  for (int i = 0; i < 16; ++i) b[i] = r[i];
  return b;
}

// ---------------------------------------------------------------------------
// fp32 -> bf16 conversion (vectorized x4)
// ---------------------------------------------------------------------------
__global__ void __launch_bounds__(256)
cast_f32_bf16_kernel(const float* __restrict__ src, bf16_t* __restrict__ dst, int n4) {
  int i = blockIdx.x * blockDim.x + threadIdx.x;
  int stride = gridDim.x * blockDim.x;
  const float4* s4 = reinterpret_cast<const float4*>(src);
  v4bf* d4 = reinterpret_cast<v4bf*>(dst);
  for (; i < n4; i += stride) {
    float4 f = s4[i];
    v4bf o = {(__bf16)f.x, (__bf16)f.y, (__bf16)f.z, (__bf16)f.w};
    d4[i] = o;
  }
}

// ---------------------------------------------------------------------------
// Kernel 1: qkv = x @ W_attn + b_attn, fused RoPE on q/k, scatter to
// Q/K/V in [B][H][T][D] bf16.  128x128 block tile, Kb=32, 8 waves:
// wave (wm,wn) owns a 32x64 subtile = 2x4 WMMA fragments.
// Each 128-wide N tile is exactly one head of one of {q,k,v}.
// ---------------------------------------------------------------------------
struct QkvSmem {
  union {
    struct {
      bf16_t xs[128 * 32];  // x tile, row-major [m][k]
      bf16_t wT[128 * 32];  // W tile, N-major  [n][k]
    } stage;
    float ot[32 * 128];     // epilogue staging, 32 rows at a time
  };
};

__global__ void __launch_bounds__(256)
qkv_rope_kernel(const bf16_t* __restrict__ xB, const bf16_t* __restrict__ wB,
                const float* __restrict__ bias,
                bf16_t* __restrict__ Qo, bf16_t* __restrict__ Ko,
                bf16_t* __restrict__ Vo) {
  __shared__ QkvSmem sm;
  const int tid  = threadIdx.x;
  const int lane = tid & 31;
  const int wid  = tid >> 5;
  const int wm   = wid & 3;   // 4 wave-rows of 32
  const int wn   = wid >> 2;  // 2 wave-cols of 64
  const int n0   = blockIdx.x * 128;
  const int m0   = blockIdx.y * 128;

  const v8f vzero = {0, 0, 0, 0, 0, 0, 0, 0};
  v8f acc[2][4];
#pragma unroll
  for (int i = 0; i < 2; ++i)
#pragma unroll
    for (int j = 0; j < 4; ++j) acc[i][j] = vzero;

  const int wk = tid >> 3, wnb = (tid & 7) * 16;  // W-stage mapping

  for (int kb = 0; kb < CC; kb += 32) {
    __syncthreads();
    // stage x tile [128 m][32 k]: async global->LDS, 512 x 16B chunks
#pragma unroll
    for (int i = 0; i < 2; ++i) {
      int c   = tid + i * 256;
      int row = c >> 2;            // 4 chunks (64B) per row
      int cb  = (c & 3) * 16;      // byte offset within row
      async_ld_b128((char*)sm.stage.xs + row * 64 + cb,
                    (const char*)(xB + (size_t)(m0 + row) * CC + kb) + cb);
    }
    // stage W tile transposed -> wT[n][k] (manual: transpose in flight)
    {
      const bf16_t* g = wB + (size_t)(kb + wk) * NQKV + n0 + wnb;
#pragma unroll
      for (int i = 0; i < 16; ++i) sm.stage.wT[(wnb + i) * 32 + wk] = g[i];
    }
    if (kb + 32 < CC) {  // hint next W tile toward L2 (global_prefetch_b8)
      __builtin_prefetch(wB + (size_t)(kb + 32 + wk) * NQKV + n0 + wnb, 0, 1);
    }
    WAIT_ASYNC0();
    __syncthreads();

    v16bf a0 = load_a_frag(&sm.stage.xs[(wm * 32 + 0) * 32], 32);
    v16bf a1 = load_a_frag(&sm.stage.xs[(wm * 32 + 16) * 32], 32);
#pragma unroll
    for (int nf = 0; nf < 4; ++nf) {
      v16bf b = load_b_fragT(&sm.stage.wT[(wn * 64 + nf * 16) * 32], 32);
      acc[0][nf] = wmma_bf16f32(a0, b, acc[0][nf]);
      acc[1][nf] = wmma_bf16f32(a1, b, acc[1][nf]);
    }
  }

  // ---- epilogue: bias + RoPE, 32 output rows per pass through LDS ----
  const int sec = n0 / CC;                // 0=q, 1=k, 2=v
  const int hd  = (n0 % CC) / DD;         // head
  const float LOG1E4_64 = 0.14391156f;    // ln(10000)/64

  for (int wmSel = 0; wmSel < 4; ++wmSel) {
    __syncthreads();
    if (wm == wmSel) {
#pragma unroll
      for (int mf = 0; mf < 2; ++mf)
#pragma unroll
        for (int nf = 0; nf < 4; ++nf) {
          int r = mf * 16 + ((lane >> 4) << 3);
          int c = wn * 64 + nf * 16 + (lane & 15);
          float bv = bias[n0 + c];
#pragma unroll
          for (int v = 0; v < 8; ++v)
            sm.ot[(r + v) * 128 + c] = acc[mf][nf][v] + bv;
        }
    }
    __syncthreads();
    const int rowBase = wmSel * 32;
    if (sec == 2) {
      // V: straight convert + scatter to [B][H][T][D]
#pragma unroll 4
      for (int i = 0; i < 16; ++i) {
        int idx = tid + i * 256;               // 32*128 elements
        int r = idx >> 7, d = idx & 127;
        int m = m0 + rowBase + r;
        int b = m >> 11, t = m & (TT - 1);
        Vo[(((size_t)(b * HH + hd) * TT) + t) * DD + d] = (bf16_t)sm.ot[r * 128 + d];
      }
    } else {
      bf16_t* dst = (sec == 0) ? Qo : Ko;
#pragma unroll 2
      for (int i = 0; i < 8; ++i) {
        int p = tid + i * 256;                 // 32*64 rope pairs
        int r = p >> 6, d2 = p & 63;
        int m = m0 + rowBase + r;
        int b = m >> 11, t = m & (TT - 1);
        float inv = __expf(-(float)d2 * LOG1E4_64);  // 10000^(-d2/64)
        float sn, cs;
        __sincosf((float)t * inv, &sn, &cs);
        float x1 = sm.ot[r * 128 + 2 * d2];
        float x2 = sm.ot[r * 128 + 2 * d2 + 1];
        size_t base = (((size_t)(b * HH + hd) * TT) + t) * DD + 2 * d2;
        dst[base]     = (bf16_t)(x1 * cs - x2 * sn);
        dst[base + 1] = (bf16_t)(x1 * sn + x2 * cs);
      }
    }
  }
}

// ---------------------------------------------------------------------------
// Kernel 2: causal flash attention per (q-tile of 128, b*H).
// 8 waves; each wave owns 16 q rows. KV streamed in tiles of 32.
// ---------------------------------------------------------------------------
struct FlashSmem {
  bf16_t Qs[128 * 128];   // Q tile, row-major [q][d]      32 KB
  bf16_t Ks[32 * 128];    // K tile, row-major [k][d]       8 KB
  bf16_t VsT[128 * 32];   // V tile transposed [d][k]       8 KB
  bf16_t Ps[8][16 * 32];  // per-wave P staging [q][k]      8 KB
};

__global__ void __launch_bounds__(256)
flash_attn_kernel(const bf16_t* __restrict__ Q, const bf16_t* __restrict__ K,
                  const bf16_t* __restrict__ V, bf16_t* __restrict__ Oattn) {
  __shared__ FlashSmem sm;
  const int tid  = threadIdx.x;
  const int lane = tid & 31;
  const int wid  = tid >> 5;
  const int q0   = blockIdx.x * 128;
  const int bh   = blockIdx.y;          // b*H + h
  const int b    = bh >> 4;
  const int h    = bh & 15;

  const size_t headOff = (size_t)bh * TT * DD;
  const bf16_t* Kh = K + headOff;
  const bf16_t* Vh = V + headOff;

  // stage Q tile (128x128 bf16 = 2048 x 16B) async global->LDS
  {
    const char* qg = (const char*)(Q + headOff + (size_t)q0 * DD);
    char* qs = (char*)sm.Qs;
#pragma unroll
    for (int i = 0; i < 8; ++i) {
      int c = tid + i * 256;
      async_ld_b128(qs + (size_t)c * 16, qg + (size_t)c * 16);
    }
  }

  const v8f vzero = {0, 0, 0, 0, 0, 0, 0, 0};
  v8f o[8];
#pragma unroll
  for (int f = 0; f < 8; ++f) o[f] = vzero;
  float mrow[8], lrow[8];
#pragma unroll
  for (int v = 0; v < 8; ++v) { mrow[v] = -INFINITY; lrow[v] = 0.0f; }

  const float scale = 0.08838834764831845f;  // 1/sqrt(128)
  const int nkv = (q0 >> 5) + 4;             // causal: keys up to q0+127

  for (int j = 0; j < nkv; ++j) {
    const int k0 = j * 32;
    __syncthreads();
    // stage K tile [32][128] (512 x 16B) async global->LDS
    {
      const char* kg = (const char*)(Kh + (size_t)k0 * DD);
      char* ks = (char*)sm.Ks;
#pragma unroll
      for (int i = 0; i < 2; ++i) {
        int c = tid + i * 256;
        async_ld_b128(ks + (size_t)c * 16, kg + (size_t)c * 16);
      }
    }
    // stage V tile transposed -> VsT[d][k] (manual transpose)
    {
      int k = tid >> 3, db = (tid & 7) * 16;
      const bf16_t* vg = Vh + (size_t)(k0 + k) * DD + db;
#pragma unroll
      for (int i = 0; i < 16; ++i) sm.VsT[(db + i) * 32 + k] = vg[i];
    }
    WAIT_ASYNC0();   // covers Q (first iteration) and this K tile
    __syncthreads();

    // S = Q * K^T for this wave's 16 q rows (two 16-key fragments)
    v8f s0 = vzero, s1 = vzero;
#pragma unroll
    for (int ds = 0; ds < 4; ++ds) {
      v16bf a  = load_a_frag(&sm.Qs[(wid * 16) * 128 + ds * 32], 128);
      v16bf b0 = load_b_fragT(&sm.Ks[0 * 128 + ds * 32], 128);
      v16bf b1 = load_b_fragT(&sm.Ks[16 * 128 + ds * 32], 128);
      s0 = wmma_bf16f32(a, b0, s0);
      s1 = wmma_bf16f32(a, b1, s1);
    }

    // scale + causal mask
    const int qrBase = q0 + wid * 16 + ((lane >> 4) << 3);
    const bool needMask = (k0 + 31 > q0 + wid * 16);
#pragma unroll
    for (int v = 0; v < 8; ++v) {
      s0[v] *= scale;
      s1[v] *= scale;
      if (needMask) {
        int qr  = qrBase + v;
        int kc0 = k0 + (lane & 15);
        if (kc0 > qr)      s0[v] = -INFINITY;
        if (kc0 + 16 > qr) s1[v] = -INFINITY;
      }
    }

    // online softmax: rows live at vgpr v within a 16-lane group
    float alpha[8];
#pragma unroll
    for (int v = 0; v < 8; ++v) {
      float mx = fmaxf(s0[v], s1[v]);
#pragma unroll
      for (int off = 8; off >= 1; off >>= 1)
        mx = fmaxf(mx, __shfl_xor(mx, off, 32));
      float mnew = fmaxf(mrow[v], mx);
      alpha[v] = __expf(mrow[v] - mnew);
      float p0 = __expf(s0[v] - mnew);
      float p1 = __expf(s1[v] - mnew);
      s0[v] = p0; s1[v] = p1;
      float sum = p0 + p1;
#pragma unroll
      for (int off = 8; off >= 1; off >>= 1)
        sum += __shfl_xor(sum, off, 32);
      lrow[v] = lrow[v] * alpha[v] + sum;
      mrow[v] = mnew;
    }

    // rescale accumulator
#pragma unroll
    for (int f = 0; f < 8; ++f)
#pragma unroll
      for (int v = 0; v < 8; ++v) o[f][v] *= alpha[v];

    // stage P (C-layout -> row-major [16 q][32 k]) in wave-private LDS
    {
      int r = (lane >> 4) << 3, c = lane & 15;
      bf16_t* pw = sm.Ps[wid];
#pragma unroll
      for (int v = 0; v < 8; ++v) {
        pw[(r + v) * 32 + c]      = (bf16_t)s0[v];
        pw[(r + v) * 32 + c + 16] = (bf16_t)s1[v];
      }
    }
    __syncthreads();

    // O += P * V
    v16bf ap = load_a_frag(sm.Ps[wid], 32);
#pragma unroll
    for (int f = 0; f < 8; ++f) {
      v16bf bv = load_b_fragT(&sm.VsT[(f * 16) * 32], 32);
      o[f] = wmma_bf16f32(ap, bv, o[f]);
    }
  }

  // normalize and store to [B][T][H*D] bf16 (row-major for the proj GEMM)
  {
    const int r = (lane >> 4) << 3, c = lane & 15;
#pragma unroll
    for (int f = 0; f < 8; ++f) {
#pragma unroll
      for (int v = 0; v < 8; ++v) {
        int qr = q0 + wid * 16 + r + v;
        float val = o[f][v] / lrow[v];
        Oattn[((size_t)(b * TT + qr)) * CC + h * DD + f * 16 + c] = (bf16_t)val;
      }
    }
  }
}

// ---------------------------------------------------------------------------
// Kernel 3: out = attn @ W_proj + b_proj (fp32 out).  Same tiling as QKV.
// ---------------------------------------------------------------------------
struct ProjSmem {
  bf16_t xs[128 * 32];
  bf16_t wT[128 * 32];
};

__global__ void __launch_bounds__(256)
proj_kernel(const bf16_t* __restrict__ A, const bf16_t* __restrict__ W,
            const float* __restrict__ bias, float* __restrict__ out) {
  __shared__ ProjSmem sm;
  const int tid  = threadIdx.x;
  const int lane = tid & 31;
  const int wid  = tid >> 5;
  const int wm   = wid & 3;
  const int wn   = wid >> 2;
  const int n0   = blockIdx.x * 128;
  const int m0   = blockIdx.y * 128;

  const v8f vzero = {0, 0, 0, 0, 0, 0, 0, 0};
  v8f acc[2][4];
#pragma unroll
  for (int i = 0; i < 2; ++i)
#pragma unroll
    for (int j = 0; j < 4; ++j) acc[i][j] = vzero;

  const int wk = tid >> 3, wnb = (tid & 7) * 16;

  for (int kb = 0; kb < CC; kb += 32) {
    __syncthreads();
    // stage A tile [128 m][32 k]: async global->LDS
#pragma unroll
    for (int i = 0; i < 2; ++i) {
      int c   = tid + i * 256;
      int row = c >> 2;
      int cb  = (c & 3) * 16;
      async_ld_b128((char*)sm.xs + row * 64 + cb,
                    (const char*)(A + (size_t)(m0 + row) * CC + kb) + cb);
    }
    // stage W tile transposed -> wT[n][k]
    {
      const bf16_t* g = W + (size_t)(kb + wk) * CC + n0 + wnb;
#pragma unroll
      for (int i = 0; i < 16; ++i) sm.wT[(wnb + i) * 32 + wk] = g[i];
    }
    if (kb + 32 < CC) {
      __builtin_prefetch(W + (size_t)(kb + 32 + wk) * CC + n0 + wnb, 0, 1);
    }
    WAIT_ASYNC0();
    __syncthreads();

    v16bf a0 = load_a_frag(&sm.xs[(wm * 32 + 0) * 32], 32);
    v16bf a1 = load_a_frag(&sm.xs[(wm * 32 + 16) * 32], 32);
#pragma unroll
    for (int nf = 0; nf < 4; ++nf) {
      v16bf b = load_b_fragT(&sm.wT[(wn * 64 + nf * 16) * 32], 32);
      acc[0][nf] = wmma_bf16f32(a0, b, acc[0][nf]);
      acc[1][nf] = wmma_bf16f32(a1, b, acc[1][nf]);
    }
  }

  // epilogue: bias + direct fp32 stores
#pragma unroll
  for (int mf = 0; mf < 2; ++mf)
#pragma unroll
    for (int nf = 0; nf < 4; ++nf) {
      int r = m0 + wm * 32 + mf * 16 + ((lane >> 4) << 3);
      int c = n0 + wn * 64 + nf * 16 + (lane & 15);
      float bv = bias[c];
#pragma unroll
      for (int v = 0; v < 8; ++v)
        out[(size_t)(r + v) * CC + c] = acc[mf][nf][v] + bv;
    }
}

// ---------------------------------------------------------------------------
// Launch
// ---------------------------------------------------------------------------
extern "C" void kernel_launch(void* const* d_in, const int* in_sizes, int n_in,
                              void* d_out, int out_size, void* d_ws, size_t ws_size,
                              hipStream_t stream) {
  const float* x      = (const float*)d_in[0];  // [B,T,C]
  const float* W_attn = (const float*)d_in[1];  // [C,3C]
  const float* b_attn = (const float*)d_in[2];  // [3C]
  const float* W_proj = (const float*)d_in[3];  // [C,C]
  const float* b_proj = (const float*)d_in[4];  // [C]
  float* out = (float*)d_out;                   // [B,T,C] fp32

  // workspace carve-up (bf16 buffers, 256B aligned)
  char* ws = (char*)d_ws;
  size_t off = 0;
  auto take = [&](size_t bytes) -> char* {
    char* p = ws + off;
    off += (bytes + 255) & ~(size_t)255;
    return p;
  };
  bf16_t* xB  = (bf16_t*)take((size_t)MROWS * CC * sizeof(bf16_t));
  bf16_t* WaB = (bf16_t*)take((size_t)CC * NQKV * sizeof(bf16_t));
  bf16_t* WpB = (bf16_t*)take((size_t)CC * CC * sizeof(bf16_t));
  bf16_t* Qb  = (bf16_t*)take((size_t)BB * HH * TT * DD * sizeof(bf16_t));
  bf16_t* Kb  = (bf16_t*)take((size_t)BB * HH * TT * DD * sizeof(bf16_t));
  bf16_t* Vb  = (bf16_t*)take((size_t)BB * HH * TT * DD * sizeof(bf16_t));
  bf16_t* Ab  = (bf16_t*)take((size_t)MROWS * CC * sizeof(bf16_t));

  auto cast_launch = [&](const float* s, bf16_t* d, size_t n) {
    int n4 = (int)(n / 4);
    int blocks = (n4 + 1023) / 1024;
    cast_f32_bf16_kernel<<<blocks, 256, 0, stream>>>(s, d, n4);
  };
  cast_launch(x, xB, (size_t)MROWS * CC);
  cast_launch(W_attn, WaB, (size_t)CC * NQKV);
  cast_launch(W_proj, WpB, (size_t)CC * CC);

  // QKV GEMM + bias + RoPE + head scatter: 48 N-tiles x 32 M-tiles
  qkv_rope_kernel<<<dim3(NQKV / 128, MROWS / 128), 256, 0, stream>>>(
      xB, WaB, b_attn, Qb, Kb, Vb);

  // causal flash attention: 16 q-tiles x (B*H=32)
  flash_attn_kernel<<<dim3(TT / 128, BB * HH), 256, 0, stream>>>(Qb, Kb, Vb, Ab);

  // output projection: 16 N-tiles x 32 M-tiles, fp32 out
  proj_kernel<<<dim3(CC / 128, MROWS / 128), 256, 0, stream>>>(Ab, WpB, b_proj, out);
}
